// RNN_85117661872189
// MI455X (gfx1250) — compile-verified, matrix operand order checked by reading, not run on hardware
//
#include <hip/hip_runtime.h>
#include <hip/hip_bf16.h>

typedef __attribute__((ext_vector_type(16))) __bf16 v16bf;
typedef __attribute__((ext_vector_type(8)))  float  v8f;

#define T_STEPS 2048
#define B_DIM   128
#define I_DIM   128
#define H_DIM   512
#define O_DIM   128

static __device__ __forceinline__ unsigned short f32_to_bf16(float f) {
    unsigned u = __float_as_uint(f);
    return (unsigned short)((u + 0x7FFFu + ((u >> 16) & 1u)) >> 16);
}

// ---------------- prep: f32 -> bf16 weight conversion ----------------
__global__ void k_f32_to_bf16(const float* __restrict__ src,
                              unsigned short* __restrict__ dst, int n) {
    int i = blockIdx.x * blockDim.x + threadIdx.x;
    if (i < n) dst[i] = f32_to_bf16(src[i]);
}

// table2[i][n] = sum_k emb[i][k] * Wxh_w[n][k] + Wxh_b[n] + Whh_b[n]
// (embedding lookup + input projection collapse to a 128x512 table gather)
__global__ void k_table(const float* __restrict__ emb,
                        const float* __restrict__ wxh,
                        const float* __restrict__ bxh,
                        const float* __restrict__ bhh,
                        float* __restrict__ table2) {
    int idx = blockIdx.x * blockDim.x + threadIdx.x;   // 128*512 = 65536
    int i = idx >> 9;        // embedding row
    int n = idx & 511;       // hidden unit
    const float* e = emb + i * I_DIM;
    const float* w = wxh + n * I_DIM;
    float acc = 0.f;
#pragma unroll 4
    for (int k = 0; k < I_DIM; ++k) acc += e[k] * w[k];
    table2[idx] = acc + bxh[n] + bhh[n];
}

// ---------------- persistent recurrence + fused decode ----------------
// One workgroup (16 wave32s) on one WGP. Hidden state double-buffered in
// LDS (2 x 128x512 bf16 = 256 KB of the 320 KB WGP LDS). One barrier/step.
__global__ __launch_bounds__(512, 1)
void k_rnn(const int* __restrict__ seq,            // [T,B] int32
           const float* __restrict__ table2,       // [128,512] fused xh table
           const unsigned short* __restrict__ whh, // [512,512] bf16 (row n = B-col n)
           const unsigned short* __restrict__ dec, // [128,512] bf16
           const float* __restrict__ dec_b,        // [128]
           float* __restrict__ out,                // [T,B,O] f32
           int T) {
    __shared__ __align__(16) unsigned short hBuf[2][B_DIM][H_DIM];
    __shared__ int sIdx[2][B_DIM];

    const int tid  = threadIdx.x;
    const int wave = tid >> 5;     // 0..15
    const int lane = tid & 31;
    const int half = lane >> 4;    // K-run / M-row-half select
    const int l16  = lane & 15;

    // h0 = 0
    for (int i = tid; i < B_DIM * H_DIM; i += 512)
        (&hBuf[0][0][0])[i] = 0;
    if (tid < B_DIM) sIdx[0][tid] = seq[tid];
    __syncthreads();

    int cur = 0;
    for (int t = 0; t < T; ++t) {
        const unsigned short* hOld = &hBuf[cur][0][0];
        unsigned short*       hNew = &hBuf[cur ^ 1][0][0];
        const int* idxCur = sIdx[t & 1];

        // prefetch next step's indices into the other sIdx buffer
        if ((t + 1 < T) && tid < B_DIM)
            sIdx[(t + 1) & 1][tid] = seq[(t + 1) * B_DIM + tid];

        // ===== h_new = tanh(hOld @ Whh^T + table2[idx_t]) =====
        // M=128 (8 mtiles), N=512 (32 ntiles, 2 per wave), K=512 (16 chunks)
        for (int nt = 0; nt < 2; ++nt) {
            const int ntile = wave * 2 + nt;
            const int n = ntile * 16 + l16;            // B/C column for this lane
            v8f acc[8] = {};

#pragma unroll 4
            for (int kc = 0; kc < 16; ++kc) {
                // B fragment (32x16 bf16): lane=N, two 8-elt K-runs, contiguous
                v16bf bfrag;
                const unsigned short* bp = whh + n * H_DIM + kc * 32 + half * 8;
                ((uint4*)&bfrag)[0] = *(const uint4*)(bp);
                ((uint4*)&bfrag)[1] = *(const uint4*)(bp + 16);
#pragma unroll
                for (int mt = 0; mt < 8; ++mt) {
                    // A fragment (16x32 bf16) from LDS: lane=M row
                    v16bf afrag;
                    const unsigned short* ap =
                        hOld + (mt * 16 + l16) * H_DIM + kc * 32 + half * 8;
                    ((uint4*)&afrag)[0] = *(const uint4*)(ap);
                    ((uint4*)&afrag)[1] = *(const uint4*)(ap + 16);
                    acc[mt] = __builtin_amdgcn_wmma_f32_16x16x32_bf16(
                        false, afrag, false, bfrag, (short)0, acc[mt],
                        false, false);
                }
            }

            // bias (fused table) + hardware tanh + bf16 store of new h
#pragma unroll
            for (int mt = 0; mt < 8; ++mt) {
#pragma unroll
                for (int j = 0; j < 8; ++j) {
                    int m = mt * 16 + half * 8 + j;    // C/D row for element j
                    float v = acc[mt][j] + table2[idxCur[m] * H_DIM + n];
                    float th;
                    asm("v_tanh_f32 %0, %1" : "=v"(th) : "v"(v));
                    hNew[m * H_DIM + n] = f32_to_bf16(th);
                }
            }
        }
        __syncthreads();   // hNew complete & visible; old-h reads done

        // ===== out_t = hNew @ dec^T + dec_b  (M=128, N=128, K=512) =====
        {
            const int ntile = wave & 7;
            const int mbase = (wave >> 3) * 4;         // 0 or 4
            const int n = ntile * 16 + l16;
            v8f acc[4] = {};
#pragma unroll 4
            for (int kc = 0; kc < 16; ++kc) {
                v16bf bfrag;
                const unsigned short* bp = dec + n * H_DIM + kc * 32 + half * 8;
                ((uint4*)&bfrag)[0] = *(const uint4*)(bp);
                ((uint4*)&bfrag)[1] = *(const uint4*)(bp + 16);
#pragma unroll
                for (int i = 0; i < 4; ++i) {
                    v16bf afrag;
                    const unsigned short* ap =
                        hNew + ((mbase + i) * 16 + l16) * H_DIM + kc * 32 + half * 8;
                    ((uint4*)&afrag)[0] = *(const uint4*)(ap);
                    ((uint4*)&afrag)[1] = *(const uint4*)(ap + 16);
                    acc[i] = __builtin_amdgcn_wmma_f32_16x16x32_bf16(
                        false, afrag, false, bfrag, (short)0, acc[i],
                        false, false);
                }
            }
            float db = dec_b[n];
            float* outT = out + (size_t)t * B_DIM * O_DIM;
#pragma unroll
            for (int i = 0; i < 4; ++i)
#pragma unroll
                for (int j = 0; j < 8; ++j) {
                    int m = (mbase + i) * 16 + half * 8 + j;
                    outT[m * O_DIM + n] = acc[i][j] + db;
                }
        }
        cur ^= 1;
    }
}

extern "C" void kernel_launch(void* const* d_in, const int* in_sizes, int n_in,
                              void* d_out, int out_size, void* d_ws, size_t ws_size,
                              hipStream_t stream) {
    const int*   seq   = (const int*)d_in[0];    // [T,B]
    const float* emb   = (const float*)d_in[1];  // [I,I]
    const float* wxh_w = (const float*)d_in[2];  // [H,I]
    const float* wxh_b = (const float*)d_in[3];  // [H]
    const float* whh_w = (const float*)d_in[4];  // [H,H]
    const float* whh_b = (const float*)d_in[5];  // [H]
    const float* dec_w = (const float*)d_in[6];  // [O,H]
    const float* dec_b = (const float*)d_in[7];  // [O]
    float* out = (float*)d_out;

    // workspace carve-up (< 1 MB total)
    char* ws = (char*)d_ws;
    unsigned short* whh_bf = (unsigned short*)(ws);                    // 512 KB
    unsigned short* dec_bf = (unsigned short*)(ws + 512 * 1024);      // 128 KB
    float*          table2 = (float*)(ws + 512 * 1024 + 128 * 1024);  // 256 KB

    k_f32_to_bf16<<<(H_DIM * H_DIM + 255) / 256, 256, 0, stream>>>(whh_w, whh_bf, H_DIM * H_DIM);
    k_f32_to_bf16<<<(O_DIM * H_DIM + 255) / 256, 256, 0, stream>>>(dec_w, dec_bf, O_DIM * H_DIM);
    k_table<<<(I_DIM * H_DIM) / 256, 256, 0, stream>>>(emb, wxh_w, wxh_b, whh_b, table2);

    k_rnn<<<1, 512, 0, stream>>>(seq, table2, whh_bf, dec_bf, dec_b, out, T_STEPS);
}
